// Encoder_40510131536238
// MI455X (gfx1250) — matrix operand order repaired and checked by reading, not compiled, not run on hardware
//
#include <hip/hip_runtime.h>
#include <math.h>

#define DIMC   512
#define HEADS  8
#define NATT   4
#define BATCH  4
#define SEQ    577
#define SEQR   592        // rows of padded score tensor (37*16)
#define SEQC   608        // cols of padded score tensor (19*32), 16B-aligned stride
#define DH     64
#define SCL    0.125f     // (512/8)^-0.5
#define BN_EPS 1e-5f
#define LEAKY  0.01f
#define NBLK   256
#define NEGBIG (-1e30f)

typedef __attribute__((ext_vector_type(16))) __bf16 v16bf;
typedef __attribute__((ext_vector_type(8)))  float  v8f;
typedef unsigned short u16;
typedef unsigned int   u32;

// ---------- bf16 helpers (RNE) ----------
__device__ __forceinline__ u16 f2bf(float f) {
    u32 u = __float_as_uint(f);
    u32 r = u + 0x7FFFu + ((u >> 16) & 1u);
    return (u16)(r >> 16);
}

union BF16x16 { v16bf v; u16 u[16]; uint4 q[2]; };
union F32x8   { v8f v; float f[8]; };

__device__ __forceinline__ int imin(int a, int b) { return a < b ? a : b; }

// ---------- fp32 -> bf16 convert (row-major copy) ----------
__global__ void cvt_kernel(const float* __restrict__ src, u16* __restrict__ dst, int n) {
    int i = blockIdx.x * blockDim.x + threadIdx.x;
    if (i < n) dst[i] = f2bf(src[i]);
}

// ---------- fp32 [K,N] -> bf16 transposed [N,K] ----------
__global__ void cvtT_kernel(const float* __restrict__ src, u16* __restrict__ dst, int K, int N) {
    int i = blockIdx.x * blockDim.x + threadIdx.x;
    if (i < K * N) {
        int n = i / K, k = i % K;
        dst[i] = f2bf(src[(size_t)k * N + n]);
    }
}

// ---------- V transpose: qkv[b,n,1536] (V block) -> vT[b,h,d, jpad] bf16, pad zero ----------
__global__ void vtrans_kernel(const u16* __restrict__ qkv, u16* __restrict__ vT) {
    int idx = blockIdx.x * blockDim.x + threadIdx.x;
    const int total = BATCH * HEADS * DH * SEQC;
    if (idx >= total) return;
    int j  = idx % SEQC;
    int t  = idx / SEQC;
    int d  = t % DH;
    int bh = t / DH;
    int h  = bh % HEADS;
    int b  = bh / HEADS;
    u16 v = 0;
    if (j < SEQ) v = qkv[((size_t)(b * SEQ + j)) * 1536 + 2 * DIMC + h * DH + d];
    vT[idx] = v;
}

// ---------- generic bf16 WMMA GEMM: C[M,N] = A[M,K] * B^T[N,K] + bias ----------
template <bool OUT_BF16>
__global__ void gemm_bf16_kernel(const u16* __restrict__ A, const u16* __restrict__ BT,
                                 const float* __restrict__ bias, void* __restrict__ Cout,
                                 int M, int K, int Nn, int tilesN)
{
    int tile = blockIdx.x;
    int tm = tile / tilesN, tn = tile % tilesN;
    int lane = threadIdx.x & 31;
    int half = lane >> 4, lm = lane & 15;
    int m0 = tm * 16, n0 = tn * 16;
    int col  = n0 + lm;
    int rowc = imin(m0 + lm, M - 1);

    const u16* Ar = A  + (size_t)rowc * K;
    const u16* Br = BT + (size_t)col  * K;

    F32x8 acc;
#pragma unroll
    for (int r = 0; r < 8; ++r) acc.f[r] = 0.f;

#pragma unroll 4
    for (int k0 = 0; k0 < K; k0 += 32) {
        BF16x16 a, b;
        a.q[0] = *(const uint4*)(Ar + k0 + half * 8);
        a.q[1] = *(const uint4*)(Ar + k0 + 16 + half * 8);
        b.q[0] = *(const uint4*)(Br + k0 + half * 8);
        b.q[1] = *(const uint4*)(Br + k0 + 16 + half * 8);
        acc.v = __builtin_amdgcn_wmma_f32_16x16x32_bf16(false, a.v, false, b.v,
                                                        (short)0, acc.v, false, false);
    }
    float bv = bias[col];
#pragma unroll
    for (int r = 0; r < 8; ++r) {
        int rr = m0 + r + 8 * half;
        if (rr < M) {
            float o = acc.f[r] + bv;
            if (OUT_BF16) ((u16*)Cout)[(size_t)rr * Nn + col] = f2bf(o);
            else          ((float*)Cout)[(size_t)rr * Nn + col] = o;
        }
    }
}

// ---------- fused dots/dots1 + Atten epilogue, padded sentinel output ----------
// grid: (tilesI*tilesJ, HEADS, BATCH), block 32; writes [B,H,SEQR,SEQC], pads = NEGBIG
__global__ void dots_kernel(const u16* __restrict__ qkvx, const u16* __restrict__ qkvl,
                            float* __restrict__ dots, float* __restrict__ dots1,
                            const float* __restrict__ c1w, const float* __restrict__ c1b,
                            const float* __restrict__ bn_g, const float* __restrict__ bn_b,
                            const float* __restrict__ bn_m, const float* __restrict__ bn_v,
                            const float* __restrict__ c2w, const float* __restrict__ c2b,
                            int tilesJ)
{
    int b = blockIdx.z, h = blockIdx.y;
    int ti = blockIdx.x / tilesJ, tj = blockIdx.x % tilesJ;
    int lane = threadIdx.x & 31;
    int half = lane >> 4, lm = lane & 15;

    __shared__ float p_w0[NATT], p_w1[NATT], p_cb[NATT], p_sc[NATT], p_sh[NATT], p_c2[NATT];
    __shared__ float p_c2b;
    if (lane < NATT) {
        int o = lane, idx = h * NATT + o;
        p_w0[o] = c1w[idx * 2 + 0];
        p_w1[o] = c1w[idx * 2 + 1];
        float sc = bn_g[idx] * rsqrtf(bn_v[idx] + BN_EPS);
        p_sc[o] = sc;
        p_sh[o] = bn_b[idx] - bn_m[idx] * sc;
        p_cb[o] = c1b[idx];
        p_c2[o] = c2w[idx];
    }
    if (lane == NATT) p_c2b = c2b[h];
    __syncthreads();

    int i0 = ti * 16, j0 = tj * 16;
    int rowAc = imin(i0 + lm, SEQ - 1);   // clamped loads: garbage only in sentinel cells
    int rowBc = imin(j0 + lm, SEQ - 1);
    const int qoff = h * DH;
    const int koff = DIMC + h * DH;
    const u16* qx = qkvx + (size_t)(b * SEQ + rowAc) * 1536 + qoff;
    const u16* kx = qkvx + (size_t)(b * SEQ + rowBc) * 1536 + koff;
    const u16* ql = qkvl + (size_t)(b * SEQ + rowAc) * 1536 + qoff;
    const u16* kl = qkvl + (size_t)(b * SEQ + rowBc) * 1536 + koff;

    F32x8 accP, accQ;
#pragma unroll
    for (int r = 0; r < 8; ++r) { accP.f[r] = 0.f; accQ.f[r] = 0.f; }

#pragma unroll
    for (int k0 = 0; k0 < DH; k0 += 32) {
        BF16x16 a0, b0, a1, b1;
        a0.q[0] = *(const uint4*)(qx + k0 + half * 8);
        a0.q[1] = *(const uint4*)(qx + k0 + 16 + half * 8);
        b0.q[0] = *(const uint4*)(kx + k0 + half * 8);
        b0.q[1] = *(const uint4*)(kx + k0 + 16 + half * 8);
        a1.q[0] = *(const uint4*)(ql + k0 + half * 8);
        a1.q[1] = *(const uint4*)(ql + k0 + 16 + half * 8);
        b1.q[0] = *(const uint4*)(kl + k0 + half * 8);
        b1.q[1] = *(const uint4*)(kl + k0 + 16 + half * 8);
        accP.v = __builtin_amdgcn_wmma_f32_16x16x32_bf16(false, a0.v, false, b0.v,
                                                         (short)0, accP.v, false, false);
        accQ.v = __builtin_amdgcn_wmma_f32_16x16x32_bf16(false, a1.v, false, b1.v,
                                                         (short)0, accQ.v, false, false);
    }

    size_t dbase = ((size_t)(b * HEADS + h)) * SEQR * SEQC;
    int col = j0 + lm;
    bool colIn = col < SEQ;
#pragma unroll
    for (int r = 0; r < 8; ++r) {
        int row = i0 + r + 8 * half;
        bool in = (row < SEQ) && colIn;
        float dv  = accP.f[r] * SCL;
        float dv1 = accQ.f[r] * SCL;
        float sup = p_c2b;
#pragma unroll
        for (int o = 0; o < NATT; ++o) {
            float h1 = p_w0[o] * dv + p_w1[o] * dv1 + p_cb[o];
            h1 = h1 * p_sc[o] + p_sh[o];
            h1 = (h1 > 0.f) ? h1 : LEAKY * h1;
            sup += h1 * p_c2[o];
        }
        size_t off = dbase + (size_t)row * SEQC + col;
        dots [off] = in ? (dv  + sup) : NEGBIG;   // branchless sentinel store
        dots1[off] = in ? (dv1 + sup) : NEGBIG;
    }
}

// ---------- deterministic two-stage reductions for KL (sentinels are inert) ----------
__global__ void redmax_stage1(const float* __restrict__ dp, const float* __restrict__ dq,
                              float* __restrict__ part, int perBatch)
{
    int b = blockIdx.y;
    size_t base = (size_t)b * perBatch;
    float mp = NEGBIG, mq = NEGBIG;
    for (size_t i = (size_t)blockIdx.x * blockDim.x + threadIdx.x; i < (size_t)perBatch;
         i += (size_t)NBLK * blockDim.x) {
        mp = fmaxf(mp, dp[base + i]);
        mq = fmaxf(mq, dq[base + i]);
    }
    __shared__ float smp[256], smq[256];
    smp[threadIdx.x] = mp; smq[threadIdx.x] = mq;
    __syncthreads();
    for (int s = 128; s > 0; s >>= 1) {
        if ((int)threadIdx.x < s) {
            smp[threadIdx.x] = fmaxf(smp[threadIdx.x], smp[threadIdx.x + s]);
            smq[threadIdx.x] = fmaxf(smq[threadIdx.x], smq[threadIdx.x + s]);
        }
        __syncthreads();
    }
    if (threadIdx.x == 0) {
        part[(b * NBLK + blockIdx.x) * 2 + 0] = smp[0];
        part[(b * NBLK + blockIdx.x) * 2 + 1] = smq[0];
    }
}

__global__ void redmax_stage2(const float* __restrict__ part, float* __restrict__ scal)
{
    int b = blockIdx.x;
    float mp = NEGBIG, mq = NEGBIG;
    for (int i = threadIdx.x; i < NBLK; i += blockDim.x) {
        mp = fmaxf(mp, part[(b * NBLK + i) * 2 + 0]);
        mq = fmaxf(mq, part[(b * NBLK + i) * 2 + 1]);
    }
    __shared__ float smp[256], smq[256];
    smp[threadIdx.x] = mp; smq[threadIdx.x] = mq;
    __syncthreads();
    for (int s = 128; s > 0; s >>= 1) {
        if ((int)threadIdx.x < s) {
            smp[threadIdx.x] = fmaxf(smp[threadIdx.x], smp[threadIdx.x + s]);
            smq[threadIdx.x] = fmaxf(smq[threadIdx.x], smq[threadIdx.x + s]);
        }
        __syncthreads();
    }
    if (threadIdx.x == 0) { scal[b * 4 + 0] = smp[0]; scal[b * 4 + 1] = smq[0]; }
}

__global__ void sumexp_stage1(const float* __restrict__ dp, const float* __restrict__ dq,
                              const float* __restrict__ scal, float* __restrict__ part,
                              int perBatch)
{
    int b = blockIdx.y;
    size_t base = (size_t)b * perBatch;
    float mp = scal[b * 4 + 0], mq = scal[b * 4 + 1];
    float sp = 0.f, sq = 0.f;
    for (size_t i = (size_t)blockIdx.x * blockDim.x + threadIdx.x; i < (size_t)perBatch;
         i += (size_t)NBLK * blockDim.x) {
        sp += expf(dp[base + i] - mp);   // pads: exp(-1e30-m) == 0
        sq += expf(dq[base + i] - mq);
    }
    __shared__ float ssp[256], ssq[256];
    ssp[threadIdx.x] = sp; ssq[threadIdx.x] = sq;
    __syncthreads();
    for (int s = 128; s > 0; s >>= 1) {
        if ((int)threadIdx.x < s) {
            ssp[threadIdx.x] += ssp[threadIdx.x + s];
            ssq[threadIdx.x] += ssq[threadIdx.x + s];
        }
        __syncthreads();
    }
    if (threadIdx.x == 0) {
        part[(b * NBLK + blockIdx.x) * 2 + 0] = ssp[0];
        part[(b * NBLK + blockIdx.x) * 2 + 1] = ssq[0];
    }
}

__global__ void sumexp_stage2(const float* __restrict__ part, float* __restrict__ scal)
{
    int b = blockIdx.x;
    float sp = 0.f, sq = 0.f;
    for (int i = threadIdx.x; i < NBLK; i += blockDim.x) {
        sp += part[(b * NBLK + i) * 2 + 0];
        sq += part[(b * NBLK + i) * 2 + 1];
    }
    __shared__ float ssp[256], ssq[256];
    ssp[threadIdx.x] = sp; ssq[threadIdx.x] = sq;
    __syncthreads();
    for (int s = 128; s > 0; s >>= 1) {
        if ((int)threadIdx.x < s) {
            ssp[threadIdx.x] += ssp[threadIdx.x + s];
            ssq[threadIdx.x] += ssq[threadIdx.x + s];
        }
        __syncthreads();
    }
    if (threadIdx.x == 0) { scal[b * 4 + 2] = logf(ssp[0]); scal[b * 4 + 3] = logf(ssq[0]); }
}

__global__ void kl_stage1(const float* __restrict__ dp, const float* __restrict__ dq,
                          const float* __restrict__ scal, float* __restrict__ part,
                          int perBatch)
{
    int b = blockIdx.y;
    size_t base = (size_t)b * perBatch;
    float mp = scal[b * 4 + 0], mq = scal[b * 4 + 1];
    float lsp = scal[b * 4 + 2], lsq = scal[b * 4 + 3];
    float acc = 0.f;
    for (size_t i = (size_t)blockIdx.x * blockDim.x + threadIdx.x; i < (size_t)perBatch;
         i += (size_t)NBLK * blockDim.x) {
        float lp = dp[base + i] - mp - lsp;
        float lq = dq[base + i] - mq - lsq;
        acc += expf(lq) * (lq - lp);     // pads: 0 * finite == 0
    }
    __shared__ float s[256];
    s[threadIdx.x] = acc;
    __syncthreads();
    for (int st = 128; st > 0; st >>= 1) {
        if ((int)threadIdx.x < st) s[threadIdx.x] += s[threadIdx.x + st];
        __syncthreads();
    }
    if (threadIdx.x == 0) part[b * NBLK + blockIdx.x] = s[0];
}

__global__ void kl_stage2(const float* __restrict__ part, float* __restrict__ outLoss)
{
    float a = 0.f;
    for (int i = threadIdx.x; i < BATCH * NBLK; i += blockDim.x) a += part[i];
    __shared__ float s[256];
    s[threadIdx.x] = a;
    __syncthreads();
    for (int st = 128; st > 0; st >>= 1) {
        if ((int)threadIdx.x < st) s[threadIdx.x] += s[threadIdx.x + st];
        __syncthreads();
    }
    if (threadIdx.x == 0) outLoss[0] = s[0] / (float)BATCH;
}

// ---------- flash-style softmax * V via WMMA (padded dots; V^T zero-padded) ----------
// grid: (tilesI, HEADS, BATCH), block 32 (one wave); no clamps: pads self-neutralize
__global__ void softmax_av_kernel(const float* __restrict__ dots, const u16* __restrict__ vT,
                                  u16* __restrict__ outA)
{
    int b = blockIdx.z, h = blockIdx.y, ti = blockIdx.x;
    int lane = threadIdx.x & 31;
    int half = lane >> 4, lm = lane & 15;
    __shared__ float red[32];
    __shared__ float ssrow[16];
    size_t dbase = ((size_t)(b * HEADS + h)) * SEQR * SEQC;
    int i0 = ti * 16;

    const float* dr = dots + dbase + (size_t)(i0 + lm) * SEQC;

    // row max: 2 lanes per row (even/odd j)
    float m = NEGBIG;
    for (int j = half; j < SEQC; j += 2) m = fmaxf(m, dr[j]);
    red[lane] = m;
    __syncthreads();
    float mrow = fmaxf(red[lm], red[lm + 16]);
    __syncthreads();
    // row sum of exp (pads contribute exactly 0)
    float s = 0.f;
    for (int j = half; j < SEQC; j += 2) s += __expf(dr[j] - mrow);
    red[lane] = s;
    __syncthreads();
    if (lane < 16) ssrow[lm] = red[lm] + red[lm + 16];
    __syncthreads();

    F32x8 acc[4];
#pragma unroll
    for (int c = 0; c < 4; ++c)
#pragma unroll
        for (int r = 0; r < 8; ++r) acc[c].f[r] = 0.f;

    const u16* vbase = vT + ((size_t)(b * HEADS + h)) * DH * SEQC;

    for (int j0 = 0; j0 < SEQC; j0 += 32) {
        // A fragment: four aligned b128 loads, unconditional
        const float4* p0 = (const float4*)(dr + j0 + half * 8);
        const float4* p1 = (const float4*)(dr + j0 + 16 + half * 8);
        float4 d0 = p0[0], d1 = p0[1], d2 = p1[0], d3 = p1[1];
        BF16x16 a;
        a.u[0]  = f2bf(__expf(d0.x - mrow)); a.u[1]  = f2bf(__expf(d0.y - mrow));
        a.u[2]  = f2bf(__expf(d0.z - mrow)); a.u[3]  = f2bf(__expf(d0.w - mrow));
        a.u[4]  = f2bf(__expf(d1.x - mrow)); a.u[5]  = f2bf(__expf(d1.y - mrow));
        a.u[6]  = f2bf(__expf(d1.z - mrow)); a.u[7]  = f2bf(__expf(d1.w - mrow));
        a.u[8]  = f2bf(__expf(d2.x - mrow)); a.u[9]  = f2bf(__expf(d2.y - mrow));
        a.u[10] = f2bf(__expf(d2.z - mrow)); a.u[11] = f2bf(__expf(d2.w - mrow));
        a.u[12] = f2bf(__expf(d3.x - mrow)); a.u[13] = f2bf(__expf(d3.y - mrow));
        a.u[14] = f2bf(__expf(d3.z - mrow)); a.u[15] = f2bf(__expf(d3.w - mrow));
#pragma unroll
        for (int c = 0; c < 4; ++c) {
            BF16x16 bb;
            const u16* vr = vbase + (size_t)(c * 16 + lm) * SEQC + j0 + half * 8;
            bb.q[0] = *(const uint4*)(vr);
            bb.q[1] = *(const uint4*)(vr + 16);
            acc[c].v = __builtin_amdgcn_wmma_f32_16x16x32_bf16(false, a.v, false, bb.v,
                                                               (short)0, acc[c].v, false, false);
        }
    }

#pragma unroll
    for (int r = 0; r < 8; ++r) {
        int row = i0 + r + 8 * half;
        if (row < SEQ) {
            float inv = 1.f / ssrow[r + 8 * half];
#pragma unroll
            for (int c = 0; c < 4; ++c) {
                int col = h * DH + c * 16 + lm;
                outA[((size_t)(b * SEQ + row)) * DIMC + col] = f2bf(acc[c].f[r] * inv);
            }
        }
    }
}

// ---------- host ----------
static inline char* ws_take(char*& p, size_t bytes) {
    char* r = p;
    p += (bytes + 255) & ~(size_t)255;
    return r;
}

extern "C" void kernel_launch(void* const* d_in, const int* in_sizes, int n_in,
                              void* d_out, int out_size, void* d_ws, size_t ws_size,
                              hipStream_t stream) {
    (void)in_sizes; (void)n_in; (void)out_size; (void)ws_size;
    const float* x      = (const float*)d_in[0];
    const float* l      = (const float*)d_in[1];
    // d_in[2] = mask (unused by the module)
    const float* wqkv   = (const float*)d_in[3];
    const float* bqkv   = (const float*)d_in[4];
    const float* wqkv1  = (const float*)d_in[5];
    const float* bqkv1  = (const float*)d_in[6];
    const float* w_mlp  = (const float*)d_in[7];
    const float* b_mlp  = (const float*)d_in[8];
    const float* w_mlp1 = (const float*)d_in[9];
    const float* b_mlp1 = (const float*)d_in[10];
    const float* c1w    = (const float*)d_in[11];
    const float* c1b    = (const float*)d_in[12];
    const float* bn_g   = (const float*)d_in[13];
    const float* bn_b   = (const float*)d_in[14];
    const float* bn_m   = (const float*)d_in[15];
    const float* bn_v   = (const float*)d_in[16];
    const float* c2w    = (const float*)d_in[17];
    const float* c2b    = (const float*)d_in[18];
    float* out = (float*)d_out;

    const int MN = BATCH * SEQ;               // 2308
    const int perBatch = HEADS * SEQR * SEQC; // padded element count per batch
    const int tilesI = SEQR / 16;             // 37
    const int tilesJ = SEQC / 16;             // 38
    const int tilesM = (MN + 15) / 16;        // 145

    char* p = (char*)d_ws;
    u16*  xbf      = (u16*)ws_take(p, (size_t)MN * DIMC * 2);
    u16*  lbf      = (u16*)ws_take(p, (size_t)MN * DIMC * 2);
    u16*  wqkvT    = (u16*)ws_take(p, (size_t)DIMC * 1536 * 2);   // [1536,512]
    u16*  wqkv1T   = (u16*)ws_take(p, (size_t)DIMC * 1536 * 2);
    u16*  wmlpT    = (u16*)ws_take(p, (size_t)DIMC * DIMC * 2);   // [512,512]
    u16*  wmlp1T   = (u16*)ws_take(p, (size_t)DIMC * DIMC * 2);
    u16*  qkvx     = (u16*)ws_take(p, (size_t)MN * 1536 * 2);
    u16*  qkvl     = (u16*)ws_take(p, (size_t)MN * 1536 * 2);
    u16*  vTx      = (u16*)ws_take(p, (size_t)BATCH * HEADS * DH * SEQC * 2);
    u16*  vTl      = (u16*)ws_take(p, (size_t)BATCH * HEADS * DH * SEQC * 2);
    float* dots    = (float*)ws_take(p, (size_t)BATCH * perBatch * 4);
    float* dots1   = (float*)ws_take(p, (size_t)BATCH * perBatch * 4);
    u16*  aoutx    = (u16*)ws_take(p, (size_t)MN * DIMC * 2);
    u16*  aoutl    = (u16*)ws_take(p, (size_t)MN * DIMC * 2);
    float* part    = (float*)ws_take(p, (size_t)BATCH * NBLK * 2 * 4);
    float* scal    = (float*)ws_take(p, (size_t)BATCH * 4 * 4);

    // 1) fp32 -> bf16 (activations row-major, weights transposed)
    {
        int n1 = MN * DIMC, n2 = DIMC * 1536, n3 = DIMC * DIMC;
        cvt_kernel <<<(n1 + 255) / 256, 256, 0, stream>>>(x, xbf, n1);
        cvt_kernel <<<(n1 + 255) / 256, 256, 0, stream>>>(l, lbf, n1);
        cvtT_kernel<<<(n2 + 255) / 256, 256, 0, stream>>>(wqkv,   wqkvT,  DIMC, 1536);
        cvtT_kernel<<<(n2 + 255) / 256, 256, 0, stream>>>(wqkv1,  wqkv1T, DIMC, 1536);
        cvtT_kernel<<<(n3 + 255) / 256, 256, 0, stream>>>(w_mlp,  wmlpT,  DIMC, DIMC);
        cvtT_kernel<<<(n3 + 255) / 256, 256, 0, stream>>>(w_mlp1, wmlp1T, DIMC, DIMC);
    }

    // 2) QKV GEMMs (bf16 out)
    {
        int tilesN = 1536 / 16;  // 96
        gemm_bf16_kernel<true><<<tilesM * tilesN, 32, 0, stream>>>(
            xbf, wqkvT, bqkv, qkvx, MN, DIMC, 1536, tilesN);
        gemm_bf16_kernel<true><<<tilesM * tilesN, 32, 0, stream>>>(
            lbf, wqkv1T, bqkv1, qkvl, MN, DIMC, 1536, tilesN);
    }

    // 2b) V transposes (zero-padded to SEQC)
    {
        int nt = BATCH * HEADS * DH * SEQC;
        vtrans_kernel<<<(nt + 255) / 256, 256, 0, stream>>>(qkvx, vTx);
        vtrans_kernel<<<(nt + 255) / 256, 256, 0, stream>>>(qkvl, vTl);
    }

    // 3) fused dots/dots1 + Atten (branchless sentinel-padded stores)
    dots_kernel<<<dim3(tilesI * tilesJ, HEADS, BATCH), 32, 0, stream>>>(
        qkvx, qkvl, dots, dots1, c1w, c1b, bn_g, bn_b, bn_m, bn_v, c2w, c2b, tilesJ);

    // 4) KL loss (deterministic two-stage reductions over padded arrays)
    redmax_stage1<<<dim3(NBLK, BATCH), 256, 0, stream>>>(dots, dots1, part, perBatch);
    redmax_stage2<<<BATCH, 256, 0, stream>>>(part, scal);
    sumexp_stage1<<<dim3(NBLK, BATCH), 256, 0, stream>>>(dots, dots1, scal, part, perBatch);
    sumexp_stage2<<<BATCH, 256, 0, stream>>>(part, scal);
    kl_stage1<<<dim3(NBLK, BATCH), 256, 0, stream>>>(dots, dots1, scal, part, perBatch);
    kl_stage2<<<1, 256, 0, stream>>>(part, out + 2 * (size_t)MN * DIMC);

    // 5) softmax * V
    softmax_av_kernel<<<dim3(tilesI, HEADS, BATCH), 32, 0, stream>>>(dots,  vTx, aoutx);
    softmax_av_kernel<<<dim3(tilesI, HEADS, BATCH), 32, 0, stream>>>(dots1, vTl, aoutl);

    // 6) final MLP GEMMs (fp32 out into d_out)
    {
        int tilesN = DIMC / 16;  // 32
        gemm_bf16_kernel<false><<<tilesM * tilesN, 32, 0, stream>>>(
            aoutx, wmlpT, b_mlp, out, MN, DIMC, DIMC, tilesN);
        gemm_bf16_kernel<false><<<tilesM * tilesN, 32, 0, stream>>>(
            aoutl, wmlp1T, b_mlp1, out + (size_t)MN * DIMC, MN, DIMC, DIMC, tilesN);
    }
}
